// LinearBlock_78065325572491
// MI455X (gfx1250) — compile-verified, matrix operand order checked by reading, not run on hardware
//
#include <hip/hip_runtime.h>

// ---------------------------------------------------------------------------
// Types for CDNA5 WMMA (wave32)
// ---------------------------------------------------------------------------
typedef __bf16 bf16_t;
typedef __attribute__((ext_vector_type(8)))  __bf16 v8bf;
typedef __attribute__((ext_vector_type(16))) __bf16 v16bf;
typedef __attribute__((ext_vector_type(8)))  float  v8f;

#define EPSV 1e-5f

static constexpr int M_NODES = 100000;
static constexpr int NFEATS  = 512;
static constexpr int SND     = 256;
static constexpr int NGRAPHS = 256;

// ---------------------------------------------------------------------------
// Graph segment boundaries: starts[g] = first row with batch >= g, starts[G]=M
// ---------------------------------------------------------------------------
__global__ void k_graph_starts(const int* __restrict__ batch, int* __restrict__ starts,
                               int M, int G) {
  for (int i = blockIdx.x * blockDim.x + threadIdx.x; i < M;
       i += gridDim.x * blockDim.x) {
    int b = batch[i];
    int prev = (i == 0) ? -1 : batch[i - 1];
    for (int g = prev + 1; g <= b; ++g) starts[g] = i;
    if (i == M - 1)
      for (int g = b + 1; g <= G; ++g) starts[g] = M;
  }
}

__global__ void k_zero(float* __restrict__ p, long n) {
  for (long i = blockIdx.x * (long)blockDim.x + threadIdx.x; i < n;
       i += (long)gridDim.x * blockDim.x) p[i] = 0.f;
}

__global__ void k_w2bf(const float* __restrict__ src, bf16_t* __restrict__ dst, int n) {
  for (int i = blockIdx.x * blockDim.x + threadIdx.x; i < n;
       i += gridDim.x * blockDim.x) dst[i] = (bf16_t)src[i];
}

// ---------------------------------------------------------------------------
// Segment (or global, batch==nullptr) partial sums of x and x^2.
// ---------------------------------------------------------------------------
template <int CPT>
__global__ void k_stats_partial(const float* __restrict__ x, const int* __restrict__ batch,
                                float* __restrict__ sum, float* __restrict__ sq,
                                int M, int dim) {
  int t = threadIdx.x;
  int rpb = (M + gridDim.x - 1) / gridDim.x;
  int r0 = blockIdx.x * rpb;
  int r1 = r0 + rpb; if (r1 > M) r1 = M;
  float s[CPT], q[CPT];
#pragma unroll
  for (int j = 0; j < CPT; ++j) { s[j] = 0.f; q[j] = 0.f; }
  int curg = -1;
  for (int r = r0; r < r1; ++r) {
    int g = batch ? batch[r] : 0;
    if (g != curg) {
      if (curg >= 0) {
#pragma unroll
        for (int j = 0; j < CPT; ++j) {
          atomicAdd(&sum[(size_t)curg * dim + t + j * 256], s[j]);
          atomicAdd(&sq [(size_t)curg * dim + t + j * 256], q[j]);
          s[j] = 0.f; q[j] = 0.f;
        }
      }
      curg = g;
    }
    size_t rb = (size_t)r * dim;
#pragma unroll
    for (int j = 0; j < CPT; ++j) {
      float v = x[rb + t + j * 256];
      s[j] += v; q[j] += v * v;
    }
  }
  if (curg >= 0) {
#pragma unroll
    for (int j = 0; j < CPT; ++j) {
      atomicAdd(&sum[(size_t)curg * dim + t + j * 256], s[j]);
      atomicAdd(&sq [(size_t)curg * dim + t + j * 256], q[j]);
    }
  }
}

// meanms[g,f] = mean*ms ; scale[g,f] = w * rsqrt(var + eps)
__global__ void k_stats_finalize(const float* __restrict__ sum, const float* __restrict__ sq,
                                 const int* __restrict__ starts, const float* __restrict__ wv,
                                 const float* __restrict__ msv, float* __restrict__ meanms,
                                 float* __restrict__ scale, int G, int dim, int Mglobal) {
  int idx = blockIdx.x * blockDim.x + threadIdx.x;
  if (idx >= G * dim) return;
  int g = idx / dim, f = idx - g * dim;
  float cnt;
  if (starts) { int c = starts[g + 1] - starts[g]; cnt = (float)(c < 1 ? 1 : c); }
  else cnt = (float)Mglobal;
  float mean = sum[idx] / cnt;
  float m2   = sq[idx]  / cnt;
  float mm   = mean * msv[f];
  float var  = m2 - 2.f * mm * mean + mm * mm;
  meanms[idx] = mm;
  scale[idx]  = wv[f] * rsqrtf(var + EPSV);
}

// normalize (+optional PReLU) -> bf16 GEMM input. One block per row.
template <int CPT>
__global__ void k_normalize(const float* __restrict__ x, const int* __restrict__ batch,
                            const float* __restrict__ meanms, const float* __restrict__ scale,
                            const float* __restrict__ bias, const float* __restrict__ aptr,
                            bf16_t* __restrict__ out, int dim) {
  int r = blockIdx.x, t = threadIdx.x;
  int g = batch ? batch[r] : 0;
  bool pre = (aptr != nullptr);
  float aval = pre ? aptr[0] : 0.f;
  size_t rb = (size_t)r * dim;
  size_t gb = (size_t)g * dim;
#pragma unroll
  for (int j = 0; j < CPT; ++j) {
    int f = t + j * 256;
    float v = (x[rb + f] - meanms[gb + f]) * scale[gb + f] + bias[f];
    if (pre && v < 0.f) v *= aval;
    out[rb + f] = (bf16_t)v;
  }
}

// ---------------------------------------------------------------------------
// WMMA bf16 GEMM:  out[m,n] = sum_k A[m,k]*W[n,k] + bias[n]  (+resid)*0.5
// One wave per 32x64 C tile (2 M-tiles x 4 N-tiles of 16x16), K step 32.
// N and K are compile-time -> every load/store is base-pointer + immediate,
// avoiding 64-bit address chains and register spill.
// A-frag (16-bit 16x32): lane L -> row (L&15), K chunks k+(L>>4)*8 and
// k+16+(L>>4)*8. B mirrors with lane -> N (W rows are N-major).
// ---------------------------------------------------------------------------
__device__ __forceinline__ v16bf cat16(v8bf lo, v8bf hi) {
  return __builtin_shufflevector(lo, hi, 0, 1, 2, 3, 4, 5, 6, 7,
                                 8, 9, 10, 11, 12, 13, 14, 15);
}

template <int NTILES, int N, int K>
__global__ void k_gemm(const bf16_t* __restrict__ A, const bf16_t* __restrict__ W,
                       const float* __restrict__ bias, const float* __restrict__ resid,
                       float* __restrict__ out) {
  const int lane = threadIdx.x & 31;   // wave32: one wave per block
  const int m0 = blockIdx.x * 32;
  const int n0 = blockIdx.y * (NTILES * 16);
  const int i = lane & 15, h = lane >> 4;

  const bf16_t* __restrict__ Ab = A + (size_t)(m0 + i) * K + h * 8;
  const bf16_t* __restrict__ Bb = W + (size_t)(n0 + i) * K + h * 8;

  v8f c[2][NTILES] = {};
  for (int ks = 0; ks < K / 32; ++ks) {
    v16bf a0 = cat16(*(const v8bf*)(Ab),          *(const v8bf*)(Ab + 16));
    v16bf a1 = cat16(*(const v8bf*)(Ab + 16 * K), *(const v8bf*)(Ab + 16 * K + 16));
#pragma unroll
    for (int nt = 0; nt < NTILES; ++nt) {
      v16bf b = cat16(*(const v8bf*)(Bb + nt * 16 * K),
                      *(const v8bf*)(Bb + nt * 16 * K + 16));
      c[0][nt] = __builtin_amdgcn_wmma_f32_16x16x32_bf16(
          false, a0, false, b, (short)0, c[0][nt], false, false);
      c[1][nt] = __builtin_amdgcn_wmma_f32_16x16x32_bf16(
          false, a1, false, b, (short)0, c[1][nt], false, false);
    }
    Ab += 32;
    Bb += 32;
  }

  // C/D layout: lane (L&15) = N column, VGPR r -> M row = r + (L>>4)*8
  float* __restrict__ Ob = out + (size_t)(m0 + h * 8) * N + n0 + i;
  const float* __restrict__ Rb =
      resid ? resid + (size_t)(m0 + h * 8) * N + n0 + i : nullptr;
#pragma unroll
  for (int mt = 0; mt < 2; ++mt) {
#pragma unroll
    for (int nt = 0; nt < NTILES; ++nt) {
      float bv = bias ? bias[n0 + nt * 16 + i] : 0.f;
#pragma unroll
      for (int r = 0; r < 8; ++r) {
        int o = (mt * 16 + r) * N + nt * 16;   // compile-time * N
        float v = c[mt][nt][r] + bv;
        if (Rb) v = (v + Rb[o]) * 0.5f;
        Ob[o] = v;
      }
    }
  }
}

// ---------------------------------------------------------------------------
// Launcher
// ---------------------------------------------------------------------------
extern "C" void kernel_launch(void* const* d_in, const int* in_sizes, int n_in,
                              void* d_out, int out_size, void* d_ws, size_t ws_size,
                              hipStream_t stream) {
  (void)in_sizes; (void)n_in; (void)out_size; (void)ws_size;
  const int M = M_NODES, G = NGRAPHS;

  const float* x     = (const float*)d_in[0];
  const int*   batch = (const int*)d_in[1];
  const float* gnw[5]  = {(const float*)d_in[2],  (const float*)d_in[5],
                          (const float*)d_in[8],  (const float*)d_in[11],
                          (const float*)d_in[14]};
  const float* gnb[5]  = {(const float*)d_in[3],  (const float*)d_in[6],
                          (const float*)d_in[9],  (const float*)d_in[12],
                          (const float*)d_in[15]};
  const float* gnms[5] = {(const float*)d_in[4],  (const float*)d_in[7],
                          (const float*)d_in[10], (const float*)d_in[13],
                          (const float*)d_in[16]};
  const float* linW[5] = {(const float*)d_in[17], (const float*)d_in[19],
                          (const float*)d_in[21], (const float*)d_in[23],
                          (const float*)d_in[25]};
  const float* linb[5] = {(const float*)d_in[18], (const float*)d_in[20],
                          (const float*)d_in[22], (const float*)d_in[24],
                          (const float*)d_in[26]};
  const float* pa[4]   = {(const float*)d_in[27], (const float*)d_in[28],
                          (const float*)d_in[29], (const float*)d_in[30]};
  float* outp = (float*)d_out;

  // workspace carve-up (256B aligned)
  char* base = (char*)d_ws;
  size_t off = 0;
  auto carve = [&](size_t bytes) -> void* {
    void* p = base + off;
    off = (off + bytes + 255) & ~(size_t)255;
    return p;
  };
  bf16_t* Abuf  = (bf16_t*)carve((size_t)M * NFEATS * sizeof(bf16_t)); // 100 MB
  float*  x1    = (float*)carve((size_t)M * SND * sizeof(float));      // 100 MB
  float*  x2    = (float*)carve((size_t)M * SND * sizeof(float));      // 100 MB
  float*  hbuf  = (float*)carve((size_t)M * SND * sizeof(float));      // 100 MB (h / x3)
  float*  sumb  = (float*)carve((size_t)G * NFEATS * sizeof(float));
  float*  sqb   = (float*)carve((size_t)G * NFEATS * sizeof(float));
  float*  meanm = (float*)carve((size_t)G * NFEATS * sizeof(float));
  float*  scal  = (float*)carve((size_t)G * NFEATS * sizeof(float));
  int*    starts = (int*)carve((G + 1) * sizeof(int));
  bf16_t* Wb[5];
  const int wsz[5] = {SND * NFEATS, SND * SND, SND * SND, SND * SND, NFEATS * SND};
  for (int w = 0; w < 5; ++w) Wb[w] = (bf16_t*)carve((size_t)wsz[w] * sizeof(bf16_t));

  // 0) segment boundaries + bf16 weights
  k_graph_starts<<<256, 256, 0, stream>>>(batch, starts, M, G);
  for (int w = 0; w < 5; ++w)
    k_w2bf<<<256, 256, 0, stream>>>(linW[w], Wb[w], wsz[w]);

  auto run_stats = [&](const float* src, bool seg, int dim, int Gc, int gn) {
    k_zero<<<128, 256, 0, stream>>>(sumb, (long)Gc * dim);
    k_zero<<<128, 256, 0, stream>>>(sqb,  (long)Gc * dim);
    const int* bt = seg ? batch : nullptr;
    if (dim == NFEATS)
      k_stats_partial<2><<<2048, 256, 0, stream>>>(src, bt, sumb, sqb, M, dim);
    else
      k_stats_partial<1><<<2048, 256, 0, stream>>>(src, bt, sumb, sqb, M, dim);
    int tot = Gc * dim;
    k_stats_finalize<<<(tot + 255) / 256, 256, 0, stream>>>(
        sumb, sqb, seg ? starts : nullptr, gnw[gn], gnms[gn], meanm, scal, Gc, dim, M);
  };
  auto run_norm = [&](const float* src, bool seg, int gn, const float* ap, int dim) {
    const int* bt = seg ? batch : nullptr;
    if (dim == NFEATS)
      k_normalize<2><<<M, 256, 0, stream>>>(src, bt, meanm, scal, gnb[gn], ap, Abuf, dim);
    else
      k_normalize<1><<<M, 256, 0, stream>>>(src, bt, meanm, scal, gnb[gn], ap, Abuf, dim);
  };

  const dim3 gN256(M / 32, SND / 64);     // 3125 x 4
  const dim3 gN512(M / 32, NFEATS / 64);  // 3125 x 8

  // Stage 1: x1 = lin1(gn1(x, batch))                       [M x 512 -> M x 256]
  run_stats(x, true, NFEATS, G, 0);
  run_norm (x, true, 0, nullptr, NFEATS);
  k_gemm<4, SND, NFEATS><<<gN256, 32, 0, stream>>>(Abuf, Wb[0], linb[0], nullptr, x1);

  // Stage 2: h = lin2(prelu(gn2(x1, batch), a2))
  run_stats(x1, true, SND, G, 1);
  run_norm (x1, true, 1, pa[0], SND);
  k_gemm<4, SND, SND><<<gN256, 32, 0, stream>>>(Abuf, Wb[1], linb[1], nullptr, hbuf);

  // Stage 3: x2 = (lin3(prelu(gn3_global(h), a3)) + x1) / 2
  run_stats(hbuf, false, SND, 1, 2);
  run_norm (hbuf, false, 2, pa[1], SND);
  k_gemm<4, SND, SND><<<gN256, 32, 0, stream>>>(Abuf, Wb[2], linb[2], x1, x2);

  // Stage 4: x3 = (lin4(prelu(gn4(x2, batch), a4)) + x2) / 2
  run_stats(x2, true, SND, G, 3);
  run_norm (x2, true, 3, pa[2], SND);
  k_gemm<4, SND, SND><<<gN256, 32, 0, stream>>>(Abuf, Wb[3], linb[3], x2, hbuf);

  // Stage 5: out = lin5(prelu(gn5(x3, batch), a5))          [M x 256 -> M x 512]
  run_stats(hbuf, true, SND, G, 4);
  run_norm (hbuf, true, 4, pa[3], SND);
  k_gemm<4, NFEATS, SND><<<gN512, 32, 0, stream>>>(Abuf, Wb[4], linb[4], nullptr, outp);
}
